// TCN_21268678050154
// MI455X (gfx1250) — matrix operand order repaired and checked by reading, not compiled, not run on hardware
//
#include <hip/hip_runtime.h>
#include <math.h>

// ---------------------------------------------------------------------------
// CDNA5 WMMA (wave32): D = A(16x32 bf16) * B(32x16 bf16) + C(16x16 f32)
// ---------------------------------------------------------------------------
typedef __bf16 v16bf_t __attribute__((ext_vector_type(16)));
typedef float  v8f_t   __attribute__((ext_vector_type(8)));
typedef int    v4i_t   __attribute__((ext_vector_type(4)));

union Frag16 { v16bf_t v; unsigned short u[16]; uint4 q[2]; };

__device__ __forceinline__ unsigned short f2bf(float f) {
    unsigned u = __float_as_uint(f);
    unsigned r = u + 0x7FFFu + ((u >> 16) & 1u);
    return (unsigned short)(r >> 16);
}

#define HH   600
#define LL   1024
#define BB   16
#define NST  64
#define APITCH 48   // LDS row pitch in u16 (96B, keeps 16B alignment)

// Optional CDNA5 async global->LDS path (compile-safe probe).
// Toolchain diagnostic showed the builtin takes generic v4i* pointers.
#if defined(__has_builtin)
#  if __has_builtin(__builtin_amdgcn_global_load_async_to_lds_b128) && \
      __has_builtin(__builtin_amdgcn_s_wait_asynccnt)
#    define ASYNC_LDS 1
#  endif
#endif
#ifndef ASYNC_LDS
#  define ASYNC_LDS 0
#endif

#if ASYNC_LDS
__device__ __forceinline__ void async_cp16(const unsigned short* g, unsigned short* l) {
    __builtin_amdgcn_global_load_async_to_lds_b128((v4i_t*)g, (v4i_t*)l, 0, 0);
}
#endif

// K-group permutation so a lane-half's 16 A elements are LDS-contiguous:
// physical order [0..7, 16..23, 8..15, 24..31]
__device__ __forceinline__ int kperm(int kk) {
    int g = kk >> 3;
    g = (g == 1) ? 2 : (g == 2) ? 1 : g;
    return (g << 3) | (kk & 7);
}

__device__ __forceinline__ v16bf_t load_frag(const unsigned short* p) {
    Frag16 f;
    f.q[0] = ((const uint4*)p)[0];   // ds_load_b128
    f.q[1] = ((const uint4*)p)[1];   // ds_load_b128
    return f.v;
}

// ---------------------------------------------------------------------------
__global__ void f32_to_bf16_kernel(const float* __restrict__ in,
                                   unsigned short* __restrict__ out, long n) {
    long i = (long)blockIdx.x * 256 + threadIdx.x;
    if (i < n) out[i] = f2bf(in[i]);
}

// ---------------------------------------------------------------------------
// Generic strided WMMA GEMM over "channel x (batch,length)" activations.
//   C[m,(b,l)] = act( sum_k A[m,k]*B[k,(b,l)] + bias[m] )
// Block tile 128(M) x 64(N), 8 waves, K step 32.
// LDS: A[row][kperm(k)] (fragment-contiguous), B transposed [n][k].
// ---------------------------------------------------------------------------
__global__ __launch_bounds__(256)
void gemm_cm_kernel(const unsigned short* __restrict__ A, long sa_m, long sa_k,
                    const unsigned short* __restrict__ Bx, long sb_b, long sb_l, long sb_k,
                    const float* __restrict__ bias,
                    float* __restrict__ Cf, unsigned short* __restrict__ Cbf,
                    long sc_b, long sc_l, long sc_m,
                    int M, int K, int act) {
    __shared__ unsigned short As[128][APITCH];
    __shared__ unsigned short Bs[64][APITCH];

    const int tid  = threadIdx.x;
    const int lane = tid & 31;
    const int wave = tid >> 5;
    const int n0   = blockIdx.x * 64;
    const int b    = n0 / LL;
    const int l0   = n0 % LL;
    const int m0   = blockIdx.y * 128;
    const int mw   = wave * 16;
    const int half = lane >> 4;
    const int nl   = lane & 15;

    v8f_t acc[4];
#pragma unroll
    for (int nt = 0; nt < 4; ++nt) acc[nt] = (v8f_t)0.0f;

    for (int k0 = 0; k0 < K; k0 += 32) {
        // ---- stage A tile (128x32) ----
        if (sa_m == 1 && (sa_k & 7) == 0) {
            // contiguous over m: uint4 of 8 m's per thread
            for (int i = tid; i < 512; i += 256) {
                int kk = i >> 4, m8 = (i & 15) * 8;
                int k = k0 + kk;
                uint4 v4 = make_uint4(0, 0, 0, 0);
                if (k < K) {
                    if (m0 + m8 + 8 <= M) {
                        v4 = *(const uint4*)&A[(long)(m0 + m8) + (long)k * sa_k];
                    } else {
                        unsigned short* s = (unsigned short*)&v4;
                        for (int t = 0; t < 8; ++t) {
                            int m = m0 + m8 + t;
                            if (m < M) s[t] = A[(long)m + (long)k * sa_k];
                        }
                    }
                }
                const unsigned short* s = (const unsigned short*)&v4;
                int pc = kperm(kk);
#pragma unroll
                for (int t = 0; t < 8; ++t) As[m8 + t][pc] = s[t];
            }
        } else if (sa_k == 1 && (sa_m & 7) == 0) {
            // contiguous over k: uint4 of 8 k's; kperm keeps 8-groups contiguous
            for (int i = tid; i < 512; i += 256) {
                int mm = i >> 2, k8 = (i & 3) * 8;
                int m = m0 + mm, k = k0 + k8;
                int pc = kperm(k8);
                if (m < M && k + 8 <= K) {
#if ASYNC_LDS
                    async_cp16(&A[(long)m * sa_m + (long)k], &As[mm][pc]);
#else
                    *(uint4*)&As[mm][pc] = *(const uint4*)&A[(long)m * sa_m + (long)k];
#endif
                } else {
                    uint4 v4 = make_uint4(0, 0, 0, 0);
                    unsigned short* s = (unsigned short*)&v4;
                    if (m < M)
                        for (int t = 0; t < 8; ++t)
                            if (k + t < K) s[t] = A[(long)m * sa_m + (long)(k + t)];
                    *(uint4*)&As[mm][pc] = v4;
                }
            }
        } else {
            for (int i = tid; i < 128 * 32; i += 256) {
                int mm = i >> 5, kk = i & 31;
                int m = m0 + mm, k = k0 + kk;
                As[mm][kperm(kk)] = (m < M && k < K)
                    ? A[(long)m * sa_m + (long)k * sa_k] : (unsigned short)0;
            }
        }
        // ---- stage B tile (32x64), transposed into LDS ----
        if (sb_l == 1) {
            for (int i = tid; i < 256; i += 256) {
                int kk = i >> 3, n8 = (i & 7) * 8;
                int k = k0 + kk;
                uint4 v4 = make_uint4(0, 0, 0, 0);
                if (k < K)
                    v4 = *(const uint4*)&Bx[(long)b * sb_b + (long)(l0 + n8) + (long)k * sb_k];
                const unsigned short* s = (const unsigned short*)&v4;
#pragma unroll
                for (int t = 0; t < 8; ++t) Bs[n8 + t][kperm(kk)] = s[t];
            }
        } else {
            for (int i = tid; i < 32 * 64; i += 256) {
                int kk = i >> 6, nn = i & 63;
                int k = k0 + kk;
                Bs[nn][kperm(kk)] = (k < K)
                    ? Bx[(long)b * sb_b + (long)(l0 + nn) * sb_l + (long)k * sb_k]
                    : (unsigned short)0;
            }
        }
        if (k0 + 32 < K) {
            int pm = m0 + (tid >> 1);
            if (pm < M)
                __builtin_prefetch(&A[(long)pm * sa_m + (long)(k0 + 32) * sa_k], 0, 1);
        }
#if ASYNC_LDS
        __builtin_amdgcn_s_wait_asynccnt(0);
#endif
        __syncthreads();

        v16bf_t av = load_frag(&As[mw + nl][half * 16]);
#pragma unroll
        for (int nt = 0; nt < 4; ++nt) {
            v16bf_t bv = load_frag(&Bs[nt * 16 + nl][half * 16]);
            acc[nt] = __builtin_amdgcn_wmma_f32_16x16x32_bf16(
                false, av, false, bv, (short)0, acc[nt], false, false);
        }
        __syncthreads();
    }

#pragma unroll
    for (int nt = 0; nt < 4; ++nt) {
#pragma unroll
        for (int r = 0; r < 8; ++r) {
            int m = m0 + mw + r + 8 * half;
            if (m >= M) continue;
            int n = nt * 16 + nl;
            float v = acc[nt][r] + (bias ? bias[m] : 0.0f);
            if (act == 1) v = fmaxf(v, 0.0f);
            long addr = (long)b * sc_b + (long)(l0 + n) * sc_l + (long)m * sc_m;
            Cf[addr] = v;
            if (Cbf) Cbf[addr] = f2bf(v);
        }
    }
}

// ---------------------------------------------------------------------------
// TCN dilated causal conv (K=3) as 3 shifted WMMA GEMMs + ReLU (+residual).
// ---------------------------------------------------------------------------
__global__ __launch_bounds__(256)
void tcn_conv_kernel(const unsigned short* __restrict__ W,
                     const unsigned short* __restrict__ Xbf,
                     const float* __restrict__ bias,
                     const float* __restrict__ Res,
                     float* __restrict__ Of, unsigned short* __restrict__ Obf,
                     int dil) {
    __shared__ unsigned short As[128][APITCH];
    __shared__ unsigned short Bs[64][APITCH];

    const int tid  = threadIdx.x;
    const int lane = tid & 31;
    const int wave = tid >> 5;
    const int n0   = blockIdx.x * 64;
    const int b    = n0 / LL;
    const int t0   = n0 % LL;
    const int m0   = blockIdx.y * 128;
    const int mw   = wave * 16;
    const int half = lane >> 4;
    const int nl   = lane & 15;

    v8f_t acc[4];
#pragma unroll
    for (int nt = 0; nt < 4; ++nt) acc[nt] = (v8f_t)0.0f;

    for (int tap = 0; tap < 3; ++tap) {
        const int shift = (2 - tap) * dil;
        for (int i0 = 0; i0 < HH; i0 += 32) {
            // A: W[m, c, tap], stride 3 in c -> scalar staging
            for (int i = tid; i < 128 * 32; i += 256) {
                int mm = i >> 5, kk = i & 31;
                int m = m0 + mm, c = i0 + kk;
                As[mm][kperm(kk)] = (m < HH && c < HH)
                    ? W[(long)m * (HH * 3) + (long)c * 3 + tap]
                    : (unsigned short)0;
            }
            // B: x[b, c, t0+nn-shift], contiguous in nn -> uint4 when aligned
            for (int i = tid; i < 256; i += 256) {
                int kk = i >> 3, n8 = (i & 7) * 8;
                int c = i0 + kk;
                int tb = t0 + n8 - shift;
                uint4 v4 = make_uint4(0, 0, 0, 0);
                if (c < HH) {
                    if (tb >= 0 && (tb & 7) == 0) {
                        v4 = *(const uint4*)&Xbf[(long)b * (HH * LL) + (long)c * LL + tb];
                    } else {
                        unsigned short* s = (unsigned short*)&v4;
                        for (int t = 0; t < 8; ++t) {
                            int tt = tb + t;
                            if (tt >= 0) s[t] = Xbf[(long)b * (HH * LL) + (long)c * LL + tt];
                        }
                    }
                }
                const unsigned short* s = (const unsigned short*)&v4;
#pragma unroll
                for (int t = 0; t < 8; ++t) Bs[n8 + t][kperm(kk)] = s[t];
            }
            __syncthreads();

            v16bf_t av = load_frag(&As[mw + nl][half * 16]);
#pragma unroll
            for (int nt = 0; nt < 4; ++nt) {
                v16bf_t bv = load_frag(&Bs[nt * 16 + nl][half * 16]);
                acc[nt] = __builtin_amdgcn_wmma_f32_16x16x32_bf16(
                    false, av, false, bv, (short)0, acc[nt], false, false);
            }
            __syncthreads();
        }
    }

#pragma unroll
    for (int nt = 0; nt < 4; ++nt) {
#pragma unroll
        for (int r = 0; r < 8; ++r) {
            int m = m0 + mw + r + 8 * half;
            if (m >= HH) continue;
            int t = t0 + nt * 16 + nl;
            long addr = (long)b * (HH * LL) + (long)m * LL + t;
            float v = acc[nt][r] + bias[m];
            if (Res) v += Res[addr];
            v = fmaxf(v, 0.0f);
            Of[addr]  = v;
            Obf[addr] = f2bf(v);
        }
    }
}

// ---------------------------------------------------------------------------
// S4D prep: dtA = A*dt ; C2 = 2*C*(exp(dtA)-1)/A    (per (h,n))
// ---------------------------------------------------------------------------
__global__ void s4_prep_kernel(const float* __restrict__ log_dt,
                               const float* __restrict__ Are, const float* __restrict__ Aim,
                               const float* __restrict__ Cre, const float* __restrict__ Cim,
                               float* __restrict__ dtA, float* __restrict__ C2) {
    int i = blockIdx.x * 256 + threadIdx.x;
    if (i >= HH * NST) return;
    int h = i >> 6;
    float dt  = __expf(log_dt[h]);
    float ar  = Are[i], ai = Aim[i];
    float dre = ar * dt, dim = ai * dt;
    float er  = __expf(dre);
    float s, c;
    __sincosf(dim, &s, &c);
    float em1r = er * c - 1.0f, em1i = er * s;
    float cr = Cre[i], ci = Cim[i];
    float nr = cr * em1r - ci * em1i;
    float ni = cr * em1i + ci * em1r;
    float inv = 1.0f / (ar * ar + ai * ai);
    dtA[2 * i]     = dre;
    dtA[2 * i + 1] = dim;
    C2[2 * i]      = 2.0f * (nr * ar + ni * ai) * inv;
    C2[2 * i + 1]  = 2.0f * (ni * ar - nr * ai) * inv;
}

// ---------------------------------------------------------------------------
// SSM kernel K[h,l] = Re( sum_n C2[h,n] * exp(dtA[h,n]*l) ); grid (L/256, H)
// ---------------------------------------------------------------------------
__global__ __launch_bounds__(256)
void s4_kernel_kernel(const float* __restrict__ dtA, const float* __restrict__ C2,
                      float* __restrict__ Kf, unsigned short* __restrict__ Kbf) {
    __shared__ float sd[2 * NST];
    __shared__ float sc[2 * NST];
    int h   = blockIdx.y;
    int tid = threadIdx.x;
    if (tid < 2 * NST)       sd[tid]           = dtA[h * 2 * NST + tid];
    else if (tid < 4 * NST)  sc[tid - 2 * NST] = C2[h * 2 * NST + tid - 2 * NST];
    __syncthreads();

    int l  = blockIdx.x * 256 + tid;
    float fl = (float)l;
    float accv = 0.0f;
#pragma unroll 4
    for (int n = 0; n < NST; ++n) {
        float er = __expf(sd[2 * n] * fl);
        float s, c;
        __sincosf(sd[2 * n + 1] * fl, &s, &c);
        accv += sc[2 * n] * er * c - sc[2 * n + 1] * er * s;
    }
    long a = (long)h * LL + l;
    Kf[a]  = accv;
    Kbf[a] = f2bf(accv);
}

// ---------------------------------------------------------------------------
// S4 causal conv: Y[t,b] = sum_s K[h,t-s] X[b,h,s]  (Toeplitz WMMA per h)
// then + D[h]*x, exact GELU. grid = (L/16, H), 1 wave/block.
// ---------------------------------------------------------------------------
__global__ __launch_bounds__(32)
void s4_toeplitz_kernel(const unsigned short* __restrict__ Kbf,
                        const unsigned short* __restrict__ Xbf,
                        const float* __restrict__ Xf,
                        const float* __restrict__ Dvec,
                        float* __restrict__ Yf, unsigned short* __restrict__ Ybf) {
    __shared__ unsigned short Ts[16][APITCH];
    __shared__ unsigned short Bs[16][APITCH];
    const int h    = blockIdx.y;
    const int t0   = blockIdx.x * 16;
    const int lane = threadIdx.x;
    const int half = lane >> 4;
    const int nl   = lane & 15;

    v8f_t acc = (v8f_t)0.0f;

    for (int s0 = 0; s0 < t0 + 16; s0 += 32) {
        // Toeplitz A tile (reversed index -> scalar staging)
        for (int i = lane; i < 16 * 32; i += 32) {
            int tt = i >> 5, ss = i & 31;
            int d = t0 + tt - (s0 + ss);
            Ts[tt][kperm(ss)] = (d >= 0) ? Kbf[(long)h * LL + d] : (unsigned short)0;
        }
        // B tile transposed [b][s]: global contiguous in s -> 16B chunks
        for (int i = lane; i < 64; i += 32) {
            int nb = i >> 2, s8 = (i & 3) * 8;
            const unsigned short* gp =
                &Xbf[(long)nb * (HH * LL) + (long)h * LL + s0 + s8];
#if ASYNC_LDS
            async_cp16(gp, &Bs[nb][kperm(s8)]);
#else
            *(uint4*)&Bs[nb][kperm(s8)] = *(const uint4*)gp;
#endif
        }
#if ASYNC_LDS
        __builtin_amdgcn_s_wait_asynccnt(0);
#endif
        __syncthreads();

        v16bf_t av = load_frag(&Ts[nl][half * 16]);
        v16bf_t bv = load_frag(&Bs[nl][half * 16]);
        acc = __builtin_amdgcn_wmma_f32_16x16x32_bf16(
            false, av, false, bv, (short)0, acc, false, false);
        __syncthreads();
    }

    const float Dh = Dvec[h];
#pragma unroll
    for (int r = 0; r < 8; ++r) {
        int t = t0 + r + 8 * half;
        long addr = (long)nl * (HH * LL) + (long)h * LL + t;
        float xv = Xf[addr];
        float v  = acc[r] + Dh * xv;
        float g  = 0.5f * v * (1.0f + erff(v * 0.70710678118f));
        Yf[addr]  = g;
        Ybf[addr] = f2bf(g);
    }
}

// ---------------------------------------------------------------------------
// GLU over channels + residual + channel LayerNorm. One block per (b,l).
// ---------------------------------------------------------------------------
__global__ __launch_bounds__(256)
void s4_glu_ln_kernel(const float* __restrict__ Z,
                      const float* __restrict__ Xres,
                      const float* __restrict__ gam, const float* __restrict__ bet,
                      float* __restrict__ Xf, unsigned short* __restrict__ Xbf) {
    __shared__ float vals[HH];
    __shared__ float red[256];
    const int n   = blockIdx.x;
    const int b   = n >> 10;
    const int l   = n & (LL - 1);
    const int tid = threadIdx.x;
    const long baseZ = (long)b * (2 * HH * LL) + l;
    const long baseX = (long)b * (HH * LL) + l;

    float lsum = 0.0f;
    for (int h = tid; h < HH; h += 256) {
        float z1 = Z[baseZ + (long)h * LL];
        float z2 = Z[baseZ + (long)(h + HH) * LL];
        float v  = z1 * (1.0f / (1.0f + __expf(-z2))) + Xres[baseX + (long)h * LL];
        vals[h] = v;
        lsum += v;
    }
    red[tid] = lsum;
    __syncthreads();
    for (int s = 128; s > 0; s >>= 1) {
        if (tid < s) red[tid] += red[tid + s];
        __syncthreads();
    }
    float mu = red[0] * (1.0f / (float)HH);
    __syncthreads();

    float lvar = 0.0f;
    for (int h = tid; h < HH; h += 256) {
        float d = vals[h] - mu;
        lvar += d * d;
    }
    red[tid] = lvar;
    __syncthreads();
    for (int s = 128; s > 0; s >>= 1) {
        if (tid < s) red[tid] += red[tid + s];
        __syncthreads();
    }
    float rstd = rsqrtf(red[0] * (1.0f / (float)HH) + 1e-5f);
    __syncthreads();

    for (int h = tid; h < HH; h += 256) {
        float o = (vals[h] - mu) * rstd * gam[h] + bet[h];
        long a = baseX + (long)h * LL;
        Xf[a]  = o;
        Xbf[a] = f2bf(o);
    }
}

// ---------------------------------------------------------------------------
// Host orchestration
// ---------------------------------------------------------------------------
extern "C" void kernel_launch(void* const* d_in, const int* in_sizes, int n_in,
                              void* d_out, int out_size, void* d_ws, size_t ws_size,
                              hipStream_t stream) {
    const float* input    = (const float*)d_in[0];
    const float* W_enc    = (const float*)d_in[1];
    const float* b_enc    = (const float*)d_in[2];
    const float* tcn_w1   = (const float*)d_in[3];
    const float* tcn_b1   = (const float*)d_in[4];
    const float* tcn_w2   = (const float*)d_in[5];
    const float* tcn_b2   = (const float*)d_in[6];
    const float* s4_Wenc  = (const float*)d_in[7];
    const float* s4_benc  = (const float*)d_in[8];
    const float* s4_ldt   = (const float*)d_in[9];
    const float* s4_A_re  = (const float*)d_in[10];
    const float* s4_A_im  = (const float*)d_in[11];
    const float* s4_C_re  = (const float*)d_in[12];
    const float* s4_C_im  = (const float*)d_in[13];
    const float* s4_D     = (const float*)d_in[14];
    const float* s4_Wout  = (const float*)d_in[15];
    const float* s4_bout  = (const float*)d_in[16];
    const float* ln_g     = (const float*)d_in[17];
    const float* ln_b     = (const float*)d_in[18];
    const float* s4_Wdec  = (const float*)d_in[19];
    const float* s4_bdec  = (const float*)d_in[20];
    const float* W_dec    = (const float*)d_in[21];
    const float* b_dec    = (const float*)d_in[22];
    float* out = (float*)d_out;

    char* ws = (char*)d_ws;
    size_t off = 0;
    auto walloc = [&](size_t bytes) -> void* {
        void* p = ws + off;
        off = (off + bytes + 255) & ~(size_t)255;
        return p;
    };
    const long NBHL = (long)BB * HH * LL;
    float*          buf0_f  = (float*)walloc(NBHL * 4);
    unsigned short* buf0_bf = (unsigned short*)walloc(NBHL * 2);
    float*          buf1_f  = (float*)walloc(NBHL * 4);
    unsigned short* buf1_bf = (unsigned short*)walloc(NBHL * 2);
    float*          zf      = (float*)walloc((long)BB * 2 * HH * LL * 4);
    unsigned short* inbf    = (unsigned short*)walloc((long)BB * LL * 257 * 2);
    unsigned short* wenc_bf = (unsigned short*)walloc(257L * HH * 2);
    unsigned short* tw1_bf  = (unsigned short*)walloc(4L * HH * HH * 3 * 2);
    unsigned short* tw2_bf  = (unsigned short*)walloc(4L * HH * HH * 3 * 2);
    unsigned short* swe_bf  = (unsigned short*)walloc((long)HH * HH * 2);
    unsigned short* swo_bf  = (unsigned short*)walloc(4L * 2 * HH * HH * 2);
    unsigned short* swd_bf  = (unsigned short*)walloc((long)HH * HH * 2);
    unsigned short* wdec_bf = (unsigned short*)walloc((long)HH * 257 * 2);
    float*          dtA     = (float*)walloc((long)HH * NST * 2 * 4);
    float*          C2      = (float*)walloc((long)HH * NST * 2 * 4);
    float*          Kf      = (float*)walloc((long)HH * LL * 4);
    unsigned short* Kbf     = (unsigned short*)walloc((long)HH * LL * 2);
    (void)ws_size; (void)in_sizes; (void)n_in; (void)out_size;

    auto cvt = [&](const float* src, unsigned short* dst, long n) {
        f32_to_bf16_kernel<<<dim3((unsigned)((n + 255) / 256)), 256, 0, stream>>>(src, dst, n);
    };
    cvt(input,   inbf,    (long)BB * LL * 257);
    cvt(W_enc,   wenc_bf, 257L * HH);
    cvt(tcn_w1,  tw1_bf,  4L * HH * HH * 3);
    cvt(tcn_w2,  tw2_bf,  4L * HH * HH * 3);
    cvt(s4_Wenc, swe_bf,  (long)HH * HH);
    cvt(s4_Wout, swo_bf,  4L * 2 * HH * HH);
    cvt(s4_Wdec, swd_bf,  (long)HH * HH);
    cvt(W_dec,   wdec_bf, (long)HH * 257);

    auto gemm = [&](const unsigned short* A, long sam, long sak,
                    const unsigned short* Bx, long sbb, long sbl, long sbk,
                    const float* bias, float* Cf, unsigned short* Cbf,
                    long scb, long scl, long scm, int M, int K, int act) {
        dim3 grid((BB * LL) / 64, (M + 127) / 128);
        gemm_cm_kernel<<<grid, 256, 0, stream>>>(A, sam, sak, Bx, sbb, sbl, sbk,
                                                 bias, Cf, Cbf, scb, scl, scm, M, K, act);
    };

    const long chb = (long)HH * LL;

    // 1) encoder -> buf0 (B,H,L)
    gemm(wenc_bf, 1, HH,
         inbf, (long)LL * 257, 257, 1,
         b_enc, buf0_f, buf0_bf, chb, 1, LL, HH, 257, 0);

    // 2) TCN stack (cur = buf0)
    for (int i = 0; i < 4; ++i) {
        int dil = 1 << i;
        dim3 grid((BB * LL) / 64, (HH + 127) / 128);
        tcn_conv_kernel<<<grid, 256, 0, stream>>>(
            tw1_bf + (long)i * HH * HH * 3, buf0_bf, tcn_b1 + i * HH,
            nullptr, buf1_f, buf1_bf, dil);
        tcn_conv_kernel<<<grid, 256, 0, stream>>>(
            tw2_bf + (long)i * HH * HH * 3, buf1_bf, tcn_b2 + i * HH,
            buf0_f, buf0_f, buf0_bf, dil);
    }

    // 3) s4 encoder linear: buf0 -> buf1
    gemm(swe_bf, 1, HH,
         buf0_bf, chb, 1, LL,
         s4_benc, buf1_f, buf1_bf, chb, 1, LL, HH, HH, 0);

    // 4) S4 blocks (cur = buf1; buf0 scratch)
    for (int i = 0; i < 4; ++i) {
        s4_prep_kernel<<<dim3((HH * NST + 255) / 256), 256, 0, stream>>>(
            s4_ldt + i * HH,
            s4_A_re + (long)i * HH * NST, s4_A_im + (long)i * HH * NST,
            s4_C_re + (long)i * HH * NST, s4_C_im + (long)i * HH * NST,
            dtA, C2);
        s4_kernel_kernel<<<dim3(LL / 256, HH), 256, 0, stream>>>(dtA, C2, Kf, Kbf);
        s4_toeplitz_kernel<<<dim3(LL / 16, HH), 32, 0, stream>>>(
            Kbf, buf1_bf, buf1_f, s4_D + i * HH, buf0_f, buf0_bf);
        gemm(swo_bf + (long)i * 2 * HH * HH, HH, 1,
             buf0_bf, chb, 1, LL,
             s4_bout + i * 2 * HH, zf, nullptr,
             (long)2 * HH * LL, 1, LL, 2 * HH, HH, 0);
        s4_glu_ln_kernel<<<dim3(BB * LL), 256, 0, stream>>>(
            zf, buf1_f, ln_g + i * HH, ln_b + i * HH, buf1_f, buf1_bf);
    }

    // 5) s4 decoder linear: buf1 -> buf0
    gemm(swd_bf, 1, HH,
         buf1_bf, chb, 1, LL,
         s4_bdec, buf0_f, buf0_bf, chb, 1, LL, HH, HH, 0);

    // 6) final decoder -> out (B, L, 257)
    gemm(wdec_bf, 1, 257,
         buf0_bf, chb, 1, LL,
         b_dec, out, nullptr,
         (long)LL * 257, 257, 1, 257, HH, 0);
}
